// GRSA_23673859735853
// MI455X (gfx1250) — matrix operand order repaired and checked by reading, not compiled, not run on hardware
//
#include <hip/hip_runtime.h>
#include <hip/hip_bf16.h>

// Swin-V2 window attention block, fused for MI455X (gfx1250, wave32, WMMA).
// GEMMs: v_wmma_f32_16x16x32_f16. Weight-slab staging: global_load_async_to_lds_b128
// (ASYNCcnt path, no VGPR round-trip).

typedef _Float16 half_t;
typedef __attribute__((ext_vector_type(16))) _Float16 v16h;
typedef __attribute__((ext_vector_type(8)))  _Float16 v8h;
typedef __attribute__((ext_vector_type(8)))  float    v8f;

#define WMMA16(a, b, c) \
  __builtin_amdgcn_wmma_f32_16x16x32_f16(false, (a), false, (b), (short)0, (c), false, false)

static constexpr int SMEM_BYTES = 151584;

__device__ __forceinline__ v8f zero8() {
  v8f z;
#pragma unroll
  for (int i = 0; i < 8; ++i) z[i] = 0.f;
  return z;
}

// Async direct global->LDS copy of 16 bytes (per lane), tracked by ASYNCcnt.
__device__ __forceinline__ void async_g2l_b128(const half_t* gsrc, half_t* ldst) {
  unsigned loff = (unsigned)(size_t)(__attribute__((address_space(3))) const void*)ldst;
  asm volatile("global_load_async_to_lds_b128 %0, %1, off"
               :
               : "v"(loff), "v"(gsrc)
               : "memory");
}
__device__ __forceinline__ void wait_asynccnt0() {
  asm volatile("s_wait_asynccnt 0x0" ::: "memory");
}

// A-fragment (16x32 f16): lane l -> row (l&15), K = (l>>4)*8 .. +7 and +16..+23
__device__ __forceinline__ v16h ldA(const half_t* base, int stride, int lane) {
  const half_t* p = base + (lane & 15) * stride + ((lane >> 4) << 3);
  v8h lo = *(const v8h*)p;
  v8h hi = *(const v8h*)(p + 16);
  v16h a;
#pragma unroll
  for (int i = 0; i < 8; ++i) { a[i] = lo[i]; a[i + 8] = hi[i]; }
  return a;
}

// B-fragment (32x16 f16): lane l -> column (l&15), K = (l>>4)*8 .. +7 and +16..+23
__device__ __forceinline__ v16h ldB(const half_t* base, int stride, int lane) {
  const half_t* p = base + (lane & 15) * stride + ((lane >> 4) << 3);
  v8h lo = *(const v8h*)p;
  v8h hi = *(const v8h*)(p + 16);
  v16h b;
#pragma unroll
  for (int i = 0; i < 8; ++i) { b[i] = lo[i]; b[i + 8] = hi[i]; }
  return b;
}

// ---------------- prep kernels ----------------

// Build effective transposed f16 weights WT[p][D][C] = blockdiag(w_s) (+ I for q,k,v).
__global__ void GRSA_prep_weights(const float* __restrict__ qw, const float* __restrict__ kw,
                                  const float* __restrict__ vw, const float* __restrict__ pw,
                                  half_t* __restrict__ wt) {
  int blk = blockIdx.x;
  int p = blk >> 8, D = blk & 255, C = threadIdx.x;
  const float* w = (p == 0) ? qw : (p == 1) ? kw : (p == 2) ? vw : pw;
  int s = D >> 7;
  float val = 0.f;
  if ((C >> 7) == s) val = w[(size_t)s * 16384 + (size_t)(C & 127) * 128 + (D & 127)];
  if (p < 3 && C == D) val += 1.f;  // residual folded into q/k/v weights
  wt[(size_t)p * 65536 + (size_t)D * 256 + C] = (half_t)val;
}

// CPB-MLP table: tbl[i][h] = (relu(rpb[i] @ w1 + b1) @ w2)[h], i < 225
__global__ void GRSA_prep_cpb(const float* __restrict__ rpb, const float* __restrict__ w1,
                              const float* __restrict__ b1, const float* __restrict__ w2,
                              float* __restrict__ tbl) {
  int i = threadIdx.x;
  if (i >= 225) return;
  float t0 = rpb[i * 2], t1 = rpb[i * 2 + 1];
  float acc[8];
#pragma unroll
  for (int h = 0; h < 8; ++h) acc[h] = 0.f;
  for (int j = 0; j < 128; ++j) {
    float hv = fmaxf(t0 * w1[j] + t1 * w1[128 + j] + b1[j], 0.f);
#pragma unroll
    for (int h = 0; h < 8; ++h) acc[h] += hv * w2[j * 8 + h];
  }
#pragma unroll
  for (int h = 0; h < 8; ++h) tbl[i * 8 + h] = acc[h];
}

// bias16[h][m][t] = 16*sigmoid(tbl[rpi[m,t]][h])
__global__ void GRSA_prep_bias(const float* __restrict__ tbl, const int* __restrict__ rpi,
                               float* __restrict__ bias16) {
  int idx = blockIdx.x * 256 + threadIdx.x;  // 32768 total
  int h = idx >> 12, r = idx & 4095;
  float v = tbl[rpi[r] * 8 + h];
  bias16[idx] = 16.f / (1.f + __expf(-v));
}

// ---------------- fused per-window kernel ----------------

__global__ __launch_bounds__(256, 1)
void GRSA_swin_fused(const float* __restrict__ x, const float* __restrict__ mask,
                     const half_t* __restrict__ wt,
                     const float* __restrict__ qb, const float* __restrict__ kb,
                     const float* __restrict__ vb, const float* __restrict__ pb,
                     const float* __restrict__ logit_scale,
                     const float* __restrict__ bias16, float* __restrict__ y) {
  extern __shared__ char smem[];
  half_t* xh     = (half_t*)(smem);           // 64x256 f16  (QKV A source)
  float*  scores = (float*)(smem);            // 64x64 f32   (alias, head loop)
  half_t* attnh  = (half_t*)(smem + 16384);   // 64x64 f16   (softmax probs)
  half_t* outh   = (half_t*)(smem + 24576);   // 64x32 f16   (per-head AV out)
  half_t* qh     = (half_t*)(smem + 32768);   // 64x256 f16
  half_t* kh     = (half_t*)(smem + 65536);   // 64x256 f16
  half_t* vT     = (half_t*)(smem + 98304);   // 256x64 f16  (v transposed)
  half_t* bslab  = (half_t*)(smem + 131072);  // 256x32 f16  (B K-slab)
  float*  biasL  = (float*)(smem + 147456);   // 4x256 f32
  float*  scaleL = (float*)(smem + 151552);   // 8 f32

  const int tid  = threadIdx.x;
  const int lane = tid & 31;
  const int wave = tid >> 5;
  const int b    = blockIdx.x;
  const int m0   = (wave & 3) << 4;   // wave's 16-row strip
  const int n0w  = (wave >> 2) << 7;  // wave's 128-col half

  // ---- phase 0: biases, scales, x -> f16 ----
  biasL[tid]       = qb[tid];
  biasL[256 + tid] = kb[tid];
  biasL[512 + tid] = vb[tid];
  biasL[768 + tid] = pb[tid];
  if (tid < 8) scaleL[tid] = __expf(fminf(logit_scale[tid], 4.60517019f));  // ln(100)
  {
    const float4* xg = (const float4*)(x + (size_t)b * 16384);
#pragma unroll
    for (int i = 0; i < 16; ++i) {
      float4 v4 = xg[tid + (i << 8)];
      half_t* d = xh + ((tid + (i << 8)) << 2);
      d[0] = (half_t)v4.x; d[1] = (half_t)v4.y; d[2] = (half_t)v4.z; d[3] = (half_t)v4.w;
    }
  }
  __syncthreads();

  // ---- phase 1: Q,K,V projections (M=64,N=256,K=256), residual in weights ----
  for (int p = 0; p < 3; ++p) {
    v8f acc[8];
#pragma unroll
    for (int i = 0; i < 8; ++i) acc[i] = zero8();
    const half_t* W = wt + (size_t)p * 65536;
    for (int ks = 0; ks < 8; ++ks) {
      __syncthreads();
      {  // async-stage WT[:, ks*32 .. +32] into LDS (thread t copies its D-row)
        const half_t* src = W + ((size_t)tid << 8) + (ks << 5);
        half_t* dst = bslab + (tid << 5);
        async_g2l_b128(src,      dst);
        async_g2l_b128(src + 8,  dst + 8);
        async_g2l_b128(src + 16, dst + 16);
        async_g2l_b128(src + 24, dst + 24);
        wait_asynccnt0();
      }
      __syncthreads();
      v16h a = ldA(xh + (m0 << 8) + (ks << 5), 256, lane);
#pragma unroll
      for (int nt = 0; nt < 8; ++nt) {
        v16h bf = ldB(bslab + ((n0w + (nt << 4)) << 5), 32, lane);
        acc[nt] = WMMA16(a, bf, acc[nt]);
      }
    }
    half_t* tgt = (p == 0) ? qh : (p == 1) ? kh : vT;
#pragma unroll
    for (int nt = 0; nt < 8; ++nt) {
#pragma unroll
      for (int j = 0; j < 8; ++j) {
        int row = m0 + j + ((lane >> 4) << 3);
        int col = n0w + (nt << 4) + (lane & 15);
        float vv = acc[nt][j] + biasL[(p << 8) + col];
        if (p < 2) tgt[(row << 8) + col] = (half_t)vv;   // q,k row-major
        else       tgt[(col << 6) + row] = (half_t)vv;   // v transposed [c][t]
      }
    }
  }
  __syncthreads();

  // ---- phase 2: L2-normalize q,k per (token, head); fold logit scale into q ----
  for (int pi = tid; pi < 512; pi += 256) {
    int m = pi >> 3, h = pi & 7;
    half_t* qr = qh + (m << 8) + (h << 5);
    half_t* kr = kh + (m << 8) + (h << 5);
    float sq = 0.f, sk = 0.f;
#pragma unroll
    for (int d = 0; d < 32; ++d) {
      float qv = (float)qr[d]; sq += qv * qv;
      float kv = (float)kr[d]; sk += kv * kv;
    }
    float rq = scaleL[h] * __frsqrt_rn(fmaxf(sq, 1e-24f));
    float rk = __frsqrt_rn(fmaxf(sk, 1e-24f));
#pragma unroll
    for (int d = 0; d < 32; ++d) {
      qr[d] = (half_t)((float)qr[d] * rq);
      kr[d] = (half_t)((float)kr[d] * rk);
    }
  }
  __syncthreads();

  // ---- phase 3: per-head attention + fused P-projection accumulation ----
  const float* maskw = mask + (size_t)(b & 63) * 4096;
  const half_t* Wp = wt + (size_t)3 * 65536;
  v8f pacc[8];
#pragma unroll
  for (int i = 0; i < 8; ++i) pacc[i] = zero8();

  for (int h = 0; h < 8; ++h) {
    {  // async-stage P weight K-block for this head (consumed after 3 barriers)
      const half_t* src = Wp + ((size_t)tid << 8) + (h << 5);
      half_t* dst = bslab + (tid << 5);
      async_g2l_b128(src,      dst);
      async_g2l_b128(src + 8,  dst + 8);
      async_g2l_b128(src + 16, dst + 16);
      async_g2l_b128(src + 24, dst + 24);
    }
    const float* bh = bias16 + (h << 12);
    __builtin_prefetch(bh + (tid << 4), 0, 1);     // bias rows -> cache
    __builtin_prefetch(maskw + (tid << 4), 0, 1);  // mask rows -> cache
    // scores = (q_n * scale) @ k_n^T  (+bias +mask): 16 tiles, 2 per wave, K=32 -> 1 WMMA
#pragma unroll
    for (int t2 = 0; t2 < 2; ++t2) {
      int tI = (wave << 1) + t2, mi = tI >> 2, ni = tI & 3;
      v16h a  = ldA(qh + ((mi << 4) << 8) + (h << 5), 256, lane);
      v16h bf = ldB(kh + ((ni << 4) << 8) + (h << 5), 256, lane);
      v8f c = zero8();
      c = WMMA16(a, bf, c);
#pragma unroll
      for (int j = 0; j < 8; ++j) {
        int row = (mi << 4) + j + ((lane >> 4) << 3);
        int col = (ni << 4) + (lane & 15);
        scores[(row << 6) + col] = c[j] + bh[(row << 6) + col] + maskw[(row << 6) + col];
      }
    }
    wait_asynccnt0();  // this wave's slab writes complete before the barrier
    __syncthreads();
    // softmax over 64-wide rows (f32), probs -> f16
    if (tid < 64) {
      const float* sr = scores + (tid << 6);
      float mx = -3.0e38f;
      for (int t = 0; t < 64; ++t) mx = fmaxf(mx, sr[t]);
      float sum = 0.f;
      for (int t = 0; t < 64; ++t) sum += __expf(sr[t] - mx);
      float inv = 1.f / sum;
      half_t* ar = attnh + (tid << 6);
      for (int t = 0; t < 64; ++t) ar[t] = (half_t)(__expf(sr[t] - mx) * inv);
    }
    __syncthreads();
    // attn @ v: 8 tiles (4x2), 1 per wave, K=64 -> 2 WMMAs; out tile to LDS
    {
      int mi = wave >> 1, ni = wave & 1;
      v8f o = zero8();
#pragma unroll
      for (int kf = 0; kf < 2; ++kf) {
        v16h a  = ldA(attnh + ((mi << 4) << 6) + (kf << 5), 64, lane);
        v16h bf = ldB(vT + (((h << 5) + (ni << 4)) << 6) + (kf << 5), 64, lane);
        o = WMMA16(a, bf, o);
      }
#pragma unroll
      for (int j = 0; j < 8; ++j) {
        int row = (mi << 4) + j + ((lane >> 4) << 3);
        int col = (ni << 4) + (lane & 15);
        outh[(row << 5) + col] = (half_t)o[j];
      }
    }
    __syncthreads();
    // P projection: this head's output columns are exactly K-block h of P GEMM
    {
      v16h a = ldA(outh + (m0 << 5), 32, lane);
#pragma unroll
      for (int nt = 0; nt < 8; ++nt) {
        v16h bf = ldB(bslab + ((n0w + (nt << 4)) << 5), 32, lane);
        pacc[nt] = WMMA16(a, bf, pacc[nt]);
      }
    }
    __syncthreads();
  }

  // ---- epilogue: y = out @ Wp + pb (f32 to HBM) ----
  float* yb = y + (size_t)b * 16384;
#pragma unroll
  for (int nt = 0; nt < 8; ++nt) {
#pragma unroll
    for (int j = 0; j < 8; ++j) {
      int row = m0 + j + ((lane >> 4) << 3);
      int col = n0w + (nt << 4) + (lane & 15);
      yb[(row << 8) + col] = pacc[nt][j] + biasL[768 + col];
    }
  }
}

// ---------------- launch ----------------

extern "C" void kernel_launch(void* const* d_in, const int* in_sizes, int n_in,
                              void* d_out, int out_size, void* d_ws, size_t ws_size,
                              hipStream_t stream) {
  (void)in_sizes; (void)n_in; (void)out_size; (void)ws_size;
  const float* x    = (const float*)d_in[0];
  const float* mask = (const float*)d_in[1];
  const float* qw   = (const float*)d_in[2];
  const float* qb   = (const float*)d_in[3];
  const float* kw   = (const float*)d_in[4];
  const float* kb   = (const float*)d_in[5];
  const float* vw   = (const float*)d_in[6];
  const float* vb   = (const float*)d_in[7];
  const float* pw   = (const float*)d_in[8];
  const float* pb   = (const float*)d_in[9];
  const float* w1   = (const float*)d_in[10];
  const float* b1   = (const float*)d_in[11];
  const float* w2   = (const float*)d_in[12];
  const float* ls   = (const float*)d_in[13];
  const float* rpb  = (const float*)d_in[14];
  const int*   rpi  = (const int*)d_in[15];
  float* y = (float*)d_out;

  char* ws = (char*)d_ws;
  half_t* wt  = (half_t*)(ws);           // 4 * 256*256 f16 = 524288 B
  float*  tbl = (float*)(ws + 524288);   // 225*8 f32 (padded to 8192 B)
  float*  b16 = (float*)(ws + 532480);   // 8*64*64 f32 = 131072 B

  (void)hipFuncSetAttribute(reinterpret_cast<const void*>(GRSA_swin_fused),
                            hipFuncAttributeMaxDynamicSharedMemorySize, SMEM_BYTES);

  GRSA_prep_weights<<<1024, 256, 0, stream>>>(qw, kw, vw, pw, wt);
  GRSA_prep_cpb<<<1, 256, 0, stream>>>(rpb, w1, b1, w2, tbl);
  GRSA_prep_bias<<<128, 256, 0, stream>>>(tbl, rpi, b16);
  GRSA_swin_fused<<<2048, 256, SMEM_BYTES, stream>>>(x, mask, wt, qb, kb, vb, pb, ls, b16, y);
}
